// VQVAE_33139967656116
// MI455X (gfx1250) — matrix operand order repaired and checked by reading, not compiled, not run on hardware
//
#include <hip/hip_runtime.h>
#include <hip/hip_bf16.h>

// CDNA5 (gfx1250) wave32 WMMA types
typedef __attribute__((ext_vector_type(16))) __bf16 v16bf;
typedef __attribute__((ext_vector_type(8)))  __bf16 v8bf;
typedef __attribute__((ext_vector_type(8)))  float  v8f;

__device__ __forceinline__ unsigned short bf16bits(float f) {
  unsigned u = __builtin_bit_cast(unsigned, f);
  return (unsigned short)((u + 0x7FFFu + ((u >> 16) & 1u)) >> 16);  // RNE
}

// ---------------------------------------------------------------------------
// Weight prep: fp32 [Cout,Cin,K] (conv) or [Cin,Cout,K] (convT) -> bf16 GEMM
// layout [CoutPad][Kpad], zero padded. Runs once per launch (tiny).
// ---------------------------------------------------------------------------
__global__ __launch_bounds__(256) void prep_weights_kernel(
    const float* __restrict__ w, unsigned short* __restrict__ wp,
    int Cout, int Cin, int K, int Kpad, int CoutPad, int transposed)
{
  const int i = blockIdx.x * 256 + threadIdx.x;
  const int total = CoutPad * Kpad;
  if (i >= total) return;
  const int co = i / Kpad, kidx = i - co * Kpad;
  const int Kdim = Cin * K;
  float v = 0.f;
  if (co < Cout && kidx < Kdim) {
    const int ci = kidx / K, kk = kidx - ci * K;
    v = transposed ? w[((long)ci * Cout + co) * K + kk]
                   : w[(long)co * Kdim + kidx];
  }
  wp[i] = bf16bits(v);
}

__global__ __launch_bounds__(256) void f32_to_bf16_kernel(
    const float* __restrict__ in, unsigned short* __restrict__ out, long n)
{
  for (long i = blockIdx.x * 256L + threadIdx.x; i < n; i += (long)gridDim.x * 256L)
    out[i] = bf16bits(in[i]);
}

// ---------------------------------------------------------------------------
// Implicit-GEMM conv / convT.  Block = 8 waves = 64(M) x 128(N) output tile,
// grid = (Lout/128, CoutPad/64, B) -> no integer division in the kernel.
// Wave (wm,wn): wm 0..3 -> 16-row slice, wn 0..1 -> 64-col slice, 4 WMMAs per
// k-step reusing one A fragment.  K reduction tiled by 32, DOUBLE-BUFFERED
// through LDS (one barrier per step; next chunk's global loads issue before
// the current step's WMMAs so the matrix pipe hides load latency).
// B chunk gathered BRANCHLESSLY with all index math hoisted: per thread the
// (k, ci, kk, validity-parity) tuple is k-step-invariant; only the source
// position advances by a constant stride across the 16 row-groups.
// ---------------------------------------------------------------------------
template <int TK, int TS, int TP, bool TR, bool ACT, bool OUT32, bool MGUARD>
__global__ __launch_bounds__(256) void conv_wmma_kernel(
    const unsigned short* __restrict__ xbf, const unsigned short* __restrict__ wp,
    const float* __restrict__ bias, void* __restrict__ yout,
    int Cin, int Lin, int Cout, int Lout, int Kdim, int Kpad)
{
  __shared__ unsigned short sA[2][64 * 32];    // [buf][m][k]
  __shared__ unsigned short sB[2][128 * 32];   // [buf][n][k] (transposed)

  const int tid  = threadIdx.x;
  const int lane = tid & 31;
  const int wid  = tid >> 5;
  const int wm   = wid & 3;                    // wave M row
  const int wn   = wid >> 2;                   // wave N col
  const int m    = lane & 15;
  const int h    = lane >> 4;

  const int b       = blockIdx.z;
  const int co_base = blockIdx.y << 6;
  const int t_base  = blockIdx.x << 7;

  const unsigned short* xb = xbf + (long)b * Cin * Lin;

  // ---- per-thread loader constants (k-step-invariant except kidx<Kdim) ----
  const int kcol = tid & 31;                   // k within chunk
  const int n0   = tid >> 5;                   // first of 16 n-rows (step 8)
  const int rrA  = tid >> 2;                   // A chunk row
  const int c8A  = (tid & 3) << 3;             // A chunk col group

  auto load_chunk = [&](int buf, int kb) {
    // A: 64x32 from padded weights, one aligned 16B load per thread
    const uint4 va = *(const uint4*)&wp[(long)(co_base + rrA) * Kpad + kb + c8A];
    *(uint4*)&sA[buf][rrA * 32 + c8A] = va;
    if (kb + 32 < Kpad)                        // speculative prefetch next chunk
      __builtin_prefetch(&wp[(long)(co_base + rrA) * Kpad + kb + 32 + c8A], 0, 1);

    // B: branchless gather, strength-reduced
    const int kidx = kb + kcol;
    const int ci   = kidx / TK;                // TK compile-time
    const int kk   = kidx - ci * TK;
    const unsigned short* xrow = xb + (long)min(ci, Cin - 1) * Lin;
    bool vok = kidx < Kdim;
    int pos, dpos;
    if constexpr (!TR) {
      pos  = (t_base + n0) * TS + kk - TP;
      dpos = 8 * TS;
    } else if constexpr (TS == 2) {
      const int num = t_base + n0 + TP - kk;
      vok  = vok && ((num & 1) == 0);          // sign-safe parity
      pos  = num >> 1;                         // arithmetic shift
      dpos = 4;
    } else {
      pos  = t_base + n0 + TP - kk;
      dpos = 8;
    }
    unsigned short* sbp = &sB[buf][n0 * 32 + kcol];
#pragma unroll
    for (int j = 0; j < 16; ++j) {
      const bool ok = vok && (pos >= 0) && (pos < Lin);
      const int posc = min(max(pos, 0), Lin - 1);
      const unsigned short v = xrow[posc];
      sbp[j * 8 * 32] = ok ? v : (unsigned short)0;   // v_cndmask, no branches
      pos += dpos;
    }
  };

  v8f acc[4] = {{}, {}, {}, {}};
  const int nsteps = (Kdim + 31) >> 5;

  load_chunk(0, 0);
  for (int s = 0; s < nsteps; ++s) {
    __syncthreads();                           // buf[s&1] ready; buf[(s+1)&1] free
    if (s + 1 < nsteps) load_chunk((s + 1) & 1, (s + 1) << 5);

    const unsigned short* cA = sA[s & 1];
    const unsigned short* cB = sB[s & 1];

    const int mA = (wm << 4) + m;
    const v8bf a_lo = *(const v8bf*)&cA[mA * 32 + (h << 3)];
    const v8bf a_hi = *(const v8bf*)&cA[mA * 32 + 16 + (h << 3)];
    const v16bf Af = __builtin_shufflevector(a_lo, a_hi,
        0, 1, 2, 3, 4, 5, 6, 7, 8, 9, 10, 11, 12, 13, 14, 15);

#pragma unroll
    for (int nn = 0; nn < 4; ++nn) {
      const int nB = (wn << 6) + (nn << 4) + m;
      const v8bf b_lo = *(const v8bf*)&cB[nB * 32 + (h << 3)];
      const v8bf b_hi = *(const v8bf*)&cB[nB * 32 + 16 + (h << 3)];
      const v16bf Bf = __builtin_shufflevector(b_lo, b_hi,
          0, 1, 2, 3, 4, 5, 6, 7, 8, 9, 10, 11, 12, 13, 14, 15);
      acc[nn] = __builtin_amdgcn_wmma_f32_16x16x32_bf16(
          false, Af, false, Bf, (short)0, acc[nn], false, false);
    }
  }

  // ---- epilogue: bias + LeakyReLU, D row = rr + 8h, col = lane&15 ----
  float bv[8];
#pragma unroll
  for (int rr = 0; rr < 8; ++rr) {
    const int co = co_base + (wm << 4) + rr + (h << 3);
    bv[rr] = (!MGUARD || co < Cout) ? bias[min(co, Cout - 1)] : 0.f;
  }
#pragma unroll
  for (int nn = 0; nn < 4; ++nn) {
    const int tcol = t_base + (wn << 6) + (nn << 4) + m;
#pragma unroll
    for (int rr = 0; rr < 8; ++rr) {
      const int co = co_base + (wm << 4) + rr + (h << 3);
      if (!MGUARD || co < Cout) {
        float val = acc[nn][rr] + bv[rr];
        if constexpr (ACT) val = (val >= 0.f) ? val : 0.01f * val;
        const long idx = ((long)b * Cout + co) * Lout + tcol;
        if constexpr (OUT32) ((float*)yout)[idx] = val;
        else ((unsigned short*)yout)[idx] = bf16bits(val);
      }
    }
  }
}

// ---------------------------------------------------------------------------
// Vector quantizer: exact fp32 argmin over 512 codes (D=64), codebook staged
// through LDS in 32KB chunks. quantized emitted as bf16 for the decoder;
// loss partials deterministic.
// ---------------------------------------------------------------------------
#define VQ_CHUNK 128
__global__ __launch_bounds__(256) void vq_kernel(
    const float* __restrict__ qin, const float* __restrict__ cb,
    unsigned short* __restrict__ quant, float* __restrict__ loss_parts, int nvec)
{
  __shared__ float scb[VQ_CHUNK * 64];
  __shared__ float red[256];
  const int tid = threadIdx.x;
  const int vid = blockIdx.x * 256 + tid;
  const bool valid = vid < nvec;

  float xv[64];
  if (valid) {
    const float* xp = qin + (long)vid * 64;
#pragma unroll
    for (int d = 0; d < 64; ++d) xv[d] = xp[d];
  }

  float best = 3.402823466e38f;
  int   bidx = 0;
  for (int c0 = 0; c0 < 512; c0 += VQ_CHUNK) {
    __syncthreads();
    for (int i = tid; i < VQ_CHUNK * 64; i += 256) scb[i] = cb[(long)c0 * 64 + i];
    __syncthreads();
    if (valid) {
      for (int k = 0; k < VQ_CHUNK; ++k) {
        const float* c = &scb[k * 64];
        float s = 0.f;
#pragma unroll
        for (int d = 0; d < 64; ++d) { float df = xv[d] - c[d]; s = fmaf(df, df, s); }
        if (s < best) { best = s; bidx = c0 + k; }   // strict < keeps first min
      }
    }
  }

  float lsum = 0.f;
  if (valid) {
    const float* c = cb + (long)bidx * 64;
    unsigned short* qp = quant + (long)vid * 64;
#pragma unroll
    for (int d = 0; d < 64; ++d) {
      const float cv = c[d];
      qp[d] = bf16bits(cv);
      const float df = cv - xv[d];
      lsum = fmaf(df, df, lsum);
    }
  }
  red[tid] = lsum;
  __syncthreads();
  for (int off = 128; off > 0; off >>= 1) {
    if (tid < off) red[tid] += red[tid + off];
    __syncthreads();
  }
  if (tid == 0) loss_parts[blockIdx.x] = red[0];
}

__global__ __launch_bounds__(256) void finalize_loss_kernel(
    const float* __restrict__ parts, float* __restrict__ out, float scale, int n)
{
  __shared__ float red[256];
  float v = 0.f;
  for (int i = threadIdx.x; i < n; i += 256) v += parts[i];
  red[threadIdx.x] = v;
  __syncthreads();
  for (int off = 128; off > 0; off >>= 1) {
    if (threadIdx.x < off) red[threadIdx.x] += red[threadIdx.x + off];
    __syncthreads();
  }
  if (threadIdx.x == 0) *out = red[0] * scale;
}

// ---------------------------------------------------------------------------
template <int TK, int TS, int TP, bool TR, bool ACT, bool OUT32, bool MGUARD>
static inline void run_conv(const unsigned short* x, const unsigned short* wp,
                            const float* bias, void* y,
                            int B, int Cin, int Lin, int Cout, int Lout,
                            int Kdim, int Kpad, int CoutPad, hipStream_t stream)
{
  dim3 grid((unsigned)(Lout >> 7), (unsigned)(CoutPad >> 6), (unsigned)B);
  conv_wmma_kernel<TK, TS, TP, TR, ACT, OUT32, MGUARD>
      <<<grid, dim3(256), 0, stream>>>(
          x, wp, bias, y, Cin, Lin, Cout, Lout, Kdim, Kpad);
}

static inline void run_prep(const float* w, unsigned short* wp,
                            int Cout, int Cin, int K, int Kpad, int CoutPad,
                            int transposed, hipStream_t stream)
{
  const int total = CoutPad * Kpad;
  prep_weights_kernel<<<dim3((total + 255) / 256), dim3(256), 0, stream>>>(
      w, wp, Cout, Cin, K, Kpad, CoutPad, transposed);
}

extern "C" void kernel_launch(void* const* d_in, const int* in_sizes, int n_in,
                              void* d_out, int out_size, void* d_ws, size_t ws_size,
                              hipStream_t stream)
{
  (void)in_sizes; (void)n_in; (void)out_size;

  const float* x   = (const float*)d_in[0];
  const float* w1  = (const float*)d_in[1];   const float* b1  = (const float*)d_in[2];
  const float* w2  = (const float*)d_in[3];   const float* b2  = (const float*)d_in[4];
  const float* w3  = (const float*)d_in[5];   const float* b3  = (const float*)d_in[6];
  const float* w4  = (const float*)d_in[7];   const float* b4  = (const float*)d_in[8];
  const float* w5  = (const float*)d_in[9];   const float* b5  = (const float*)d_in[10];
  const float* qw  = (const float*)d_in[11];  const float* qb  = (const float*)d_in[12];
  const float* cbk = (const float*)d_in[13];
  const float* dw1 = (const float*)d_in[14];  const float* db1 = (const float*)d_in[15];
  const float* dw2 = (const float*)d_in[16];  const float* db2 = (const float*)d_in[17];
  const float* dw3 = (const float*)d_in[18];  const float* db3 = (const float*)d_in[19];
  const float* dw4 = (const float*)d_in[20];  const float* db4 = (const float*)d_in[21];
  const float* dw5 = (const float*)d_in[22];  const float* db5 = (const float*)d_in[23];

  // ---- workspace layout (elements are ushort unless noted) ----
  unsigned short* us = (unsigned short*)d_ws;
  const long BUF  = 33554432L;   // 64*128*4096 bf16 (largest activation)
  const long XBF  = 1048576L;    // 64*2*8192
  const long QBUF = 4194304L;    // 64*64*1024
  unsigned short* bufA  = us;                      // BUF
  unsigned short* bufB  = bufA + BUF;              // BUF
  unsigned short* xbf   = bufB + BUF;              // XBF
  unsigned short* quant = xbf + XBF;               // QBUF
  unsigned short* wbuf  = quant + QBUF;            // weights, padded bf16
  unsigned short* wp1 = wbuf;                      //  128x32
  unsigned short* wp2 = wp1 + 128L * 32;           //  256x512
  unsigned short* wp3 = wp2 + 256L * 512;          //  256x1024
  unsigned short* wp4 = wp3 + 256L * 1024;         //  256x768
  unsigned short* wp5 = wp4 + 256L * 768;          //  64x256
  unsigned short* wpq = wp5 + 64L * 256;           //  64x64
  unsigned short* wd1 = wpq + 64L * 64;            //  256x64
  unsigned short* wd2 = wd1 + 256L * 64;           //  256x768
  unsigned short* wd3 = wd2 + 256L * 768;          //  256x1024
  unsigned short* wd4 = wd3 + 256L * 1024;         //  128x1024
  unsigned short* wd5 = wd4 + 128L * 1024;         //  64x512
  unsigned short* wend = wd5 + 64L * 512;
  long ubytes = ((long)(wend - us) * 2 + 15) & ~15L;
  float* qin   = (float*)((char*)d_ws + ubytes);   // QBUF fp32 (exact VQ input)
  float* lossp = qin + QBUF;                       // 256 partials
  const long NEED = ubytes + (QBUF + 256) * 4;
  if (ws_size < (size_t)NEED) return;

  float* out = (float*)d_out;    // [64*2*8192 dec][1 latent_loss]

  // ---- one-time (per call) prep: weights -> padded bf16 GEMM layout ----
  run_prep(w1,  wp1, 128,   2, 4,   32, 128, 0, stream);
  run_prep(w2,  wp2, 256, 128, 4,  512, 256, 0, stream);
  run_prep(w3,  wp3, 256, 256, 4, 1024, 256, 0, stream);
  run_prep(w4,  wp4, 256, 256, 3,  768, 256, 0, stream);
  run_prep(w5,  wp5,  64, 256, 1,  256,  64, 0, stream);
  run_prep(qw,  wpq,  64,  64, 1,   64,  64, 0, stream);
  run_prep(dw1, wd1, 256,  64, 1,   64, 256, 1, stream);
  run_prep(dw2, wd2, 256, 256, 3,  768, 256, 1, stream);
  run_prep(dw3, wd3, 256, 256, 4, 1024, 256, 1, stream);
  run_prep(dw4, wd4, 128, 256, 4, 1024, 128, 1, stream);
  run_prep(dw5, wd5,   2, 128, 4,  512,  64, 1, stream);
  f32_to_bf16_kernel<<<dim3(1024), dim3(256), 0, stream>>>(x, xbf, XBF);

  // ----- encoder -----
  run_conv<4,2,1,false,true ,false,false>(xbf,  wp1, b1, bufA, 64,   2, 8192, 128, 4096,    8,   32, 128, stream);
  run_conv<4,2,1,false,true ,false,false>(bufA, wp2, b2, bufB, 64, 128, 4096, 256, 2048,  512,  512, 256, stream);
  run_conv<4,2,1,false,true ,false,false>(bufB, wp3, b3, bufA, 64, 256, 2048, 256, 1024, 1024, 1024, 256, stream);
  run_conv<3,1,1,false,true ,false,false>(bufA, wp4, b4, bufB, 64, 256, 1024, 256, 1024,  768,  768, 256, stream);
  run_conv<1,1,0,false,true ,false,false>(bufB, wp5, b5, bufA, 64, 256, 1024,  64, 1024,  256,  256,  64, stream);
  run_conv<1,1,0,false,false,true ,false>(bufA, wpq, qb, qin,  64,  64, 1024,  64, 1024,   64,   64,  64, stream);

  // ----- vector quantizer (65536 vectors of dim 64 vs 512 codes) -----
  vq_kernel<<<dim3(256), dim3(256), 0, stream>>>(qin, cbk, quant, lossp, 65536);

  // ----- decoder (straight-through: input = quantized) -----
  run_conv<1,1,0,true,true ,false,false>(quant, wd1, db1, bufA, 64,  64, 1024, 256, 1024,   64,   64, 256, stream);
  run_conv<3,1,1,true,true ,false,false>(bufA,  wd2, db2, bufB, 64, 256, 1024, 256, 1024,  768,  768, 256, stream);
  run_conv<4,2,1,true,true ,false,false>(bufB,  wd3, db3, bufA, 64, 256, 1024, 256, 2048, 1024, 1024, 256, stream);
  run_conv<4,2,1,true,true ,false,false>(bufA,  wd4, db4, bufB, 64, 256, 2048, 128, 4096, 1024, 1024, 128, stream);
  run_conv<4,2,1,true,false,true ,true >(bufB,  wd5, db5, out,  64, 128, 4096,   2, 8192,  512,  512,  64, stream);

  // latent_loss = q_loss + 0.25*e_loss = 1.25 * mean((quant - qin)^2)
  finalize_loss_kernel<<<dim3(1), dim3(256), 0, stream>>>(
      lossp, out + 1048576L, 1.25f / 4194304.0f, 256);
}
